// MHAToken_1125281432099
// MI455X (gfx1250) — matrix operand order repaired and checked by reading, not compiled, not run on hardware
//
#include <hip/hip_runtime.h>
#include <cstdint>
#include <cstddef>

// ---------------------------------------------------------------------------
// Types for CDNA5 WMMA (wave32): v_wmma_f32_16x16x32_bf16
// ---------------------------------------------------------------------------
typedef __attribute__((ext_vector_type(16))) __bf16 v16bf;
typedef __attribute__((ext_vector_type(8)))  float  v8f;
typedef __attribute__((ext_vector_type(4)))  int    i32x4;

#define AS1 __attribute__((address_space(1)))
#define AS3 __attribute__((address_space(3)))

#if defined(__HIP_DEVICE_COMPILE__) && defined(__gfx1250__) && \
    __has_builtin(__builtin_amdgcn_global_load_async_to_lds_b128)
#define HAVE_ASYNC_LDS 1
#else
#define HAVE_ASYNC_LDS 0
#endif

// 16-byte global -> LDS copy. Async (ASYNCcnt-tracked, no VGPR round trip)
// when the gfx1250 builtin exists; plain vector copy otherwise.
__device__ __forceinline__ void cp16(void* ldst, const void* gsrc) {
#if HAVE_ASYNC_LDS
  __builtin_amdgcn_global_load_async_to_lds_b128(
      (AS1 i32x4*)(uintptr_t)gsrc,
      (AS3 i32x4*)(unsigned)(uintptr_t)ldst, 0, 0);
#else
  *(uint4*)ldst = *(const uint4*)gsrc;
#endif
}

__device__ __forceinline__ void wait_async0() {
#if HAVE_ASYNC_LDS
#if __has_builtin(__builtin_amdgcn_s_wait_asynccnt)
  __builtin_amdgcn_s_wait_asynccnt(0);
#else
  asm volatile("s_wait_asynccnt 0x0" ::: "memory");
#endif
#endif
}

// Native f32 -> bf16 convert (RNE). gfx1250 has hardware bf16 converts, so a
// plain fptrunc lowers to a single VALU op (packable to v_cvt_pk_bf16_f32).
__device__ __forceinline__ __bf16 f2bf(float f) { return (__bf16)f; }

__device__ __forceinline__ v8f wmma_bf16(v16bf a, v16bf b, v8f c) {
  // (neg_a, A, neg_b, B, c_mod, C, reuse_a, reuse_b)
  return __builtin_amdgcn_wmma_f32_16x16x32_bf16(false, a, false, b,
                                                 (short)0, c, false, false);
}

// A fragment: 16x32 bf16 tile, row-major source with leading dim `ld` (elems).
// ISA layout: lanes 0-15 hold row M=lane, K={0..7,16..23};
//             lanes 16-31 hold row M=lane-16, K={8..15,24..31}.
__device__ __forceinline__ v16bf load_frag_a(const __bf16* base, int ld) {
  const int lane = threadIdx.x & 31;
  const int row  = lane & 15;
  const int ko   = (lane < 16) ? 0 : 8;
  const __bf16* p = base + row * ld + ko;
  v16bf a;
#pragma unroll
  for (int j = 0; j < 8; ++j) { a[j] = p[j]; a[8 + j] = p[16 + j]; }
  return a;
}

// B fragment: 32x16 bf16 (KxN). Source is K-transposed: baseT[n*ld + k].
// ISA layout: lanes 0-15 hold col N=lane, K={0..7,16..23}; lanes 16-31 K={8..15,24..31}.
__device__ __forceinline__ v16bf load_frag_b(const __bf16* baseT, int ld) {
  const int lane = threadIdx.x & 31;
  const int n  = lane & 15;
  const int kb = (lane < 16) ? 0 : 8;
  const __bf16* p = baseT + n * ld + kb;
  v16bf b;
#pragma unroll
  for (int j = 0; j < 8; ++j) { b[j] = p[j]; b[8 + j] = p[16 + j]; }
  return b;
}

// C/D 16x16 f32: VGPR r <-> M=r (lanes 0-15) / M=8+r (lanes 16-31), N=lane%16.
__device__ __forceinline__ void store_frag_c_f32(float* base, int ld, v8f c) {
  const int lane = threadIdx.x & 31;
  const int n  = lane & 15;
  const int mo = (lane < 16) ? 0 : 8;
#pragma unroll
  for (int r = 0; r < 8; ++r) base[(size_t)(mo + r) * ld + n] = c[r];
}

__device__ __forceinline__ void store_frag_c_bf16(__bf16* base, size_t ld, v8f c) {
  const int lane = threadIdx.x & 31;
  const int n  = lane & 15;
  const int mo = (lane < 16) ? 0 : 8;
#pragma unroll
  for (int r = 0; r < 8; ++r) base[(size_t)(mo + r) * ld + n] = f2bf(c[r]);
}

// ---------------------------------------------------------------------------
// Kernel 1: projection GEMM  Y_bf16[M,768] = X_f32[M,768] @ W_f32[768,768]
// Block tile 128x64, BK=32, 256 threads = 8 waves (4 M x 2 N), 2x2 frags/wave.
// f32 sources are converted to bf16 during staging (float4 vector loads).
// ---------------------------------------------------------------------------
__global__ __launch_bounds__(256) void proj_gemm_kernel(
    const float* __restrict__ X, const float* __restrict__ W,
    __bf16* __restrict__ Y) {
  __shared__ __align__(16) __bf16 lsA[128 * 32];   // A tile (row-major)
  __shared__ __align__(16) __bf16 lsWT[64 * 32];   // W tile, transposed [n][k]

  const int tid  = threadIdx.x;
  const int wave = tid >> 5;
  const int wm   = wave >> 1;            // 0..3
  const int wn   = wave & 1;             // 0..1
  const int m0   = blockIdx.x * 128;
  const int n0   = blockIdx.y * 64;

  v8f acc[2][2] = {};
  for (int kk = 0; kk < 768; kk += 32) {
#pragma unroll
    for (int i = 0; i < 4; ++i) {        // A: 1024 float4s, f32 -> bf16
      int idx = tid + i * 256;
      int r = idx >> 3, c4 = (idx & 7) * 4;
      float4 x4 = *(const float4*)&X[(size_t)(m0 + r) * 768 + kk + c4];
      __bf16* dst = &lsA[r * 32 + c4];
      dst[0] = f2bf(x4.x); dst[1] = f2bf(x4.y);
      dst[2] = f2bf(x4.z); dst[3] = f2bf(x4.w);
    }
#pragma unroll
    for (int i = 0; i < 2; ++i) {        // W: 512 float4s, transposed store
      int idx = tid + i * 256;
      int k = idx >> 4, n4 = (idx & 15) * 4;
      float4 w4 = *(const float4*)&W[(size_t)(kk + k) * 768 + n0 + n4];
      lsWT[(n4 + 0) * 32 + k] = f2bf(w4.x);
      lsWT[(n4 + 1) * 32 + k] = f2bf(w4.y);
      lsWT[(n4 + 2) * 32 + k] = f2bf(w4.z);
      lsWT[(n4 + 3) * 32 + k] = f2bf(w4.w);
    }
    if (kk + 32 < 768) {                 // global_prefetch_b8 for next tiles
      __builtin_prefetch(&X[(size_t)(m0 + (tid >> 1)) * 768 + kk + 32]);
      __builtin_prefetch(&W[(size_t)(kk + 32 + (tid >> 5)) * 768 + n0]);
    }
    __syncthreads();

    v16bf a0 = load_frag_a(&lsA[(wm * 32 + 0)  * 32], 32);
    v16bf a1 = load_frag_a(&lsA[(wm * 32 + 16) * 32], 32);
    v16bf b0 = load_frag_b(&lsWT[(wn * 32 + 0)  * 32], 32);
    v16bf b1 = load_frag_b(&lsWT[(wn * 32 + 16) * 32], 32);
    acc[0][0] = wmma_bf16(a0, b0, acc[0][0]);
    acc[0][1] = wmma_bf16(a0, b1, acc[0][1]);
    acc[1][0] = wmma_bf16(a1, b0, acc[1][0]);
    acc[1][1] = wmma_bf16(a1, b1, acc[1][1]);
    __syncthreads();
  }
#pragma unroll
  for (int i = 0; i < 2; ++i)
#pragma unroll
    for (int j = 0; j < 2; ++j)
      store_frag_c_bf16(Y + (size_t)(m0 + wm * 32 + i * 16) * 768
                          + n0 + wn * 32 + j * 16, 768, acc[i][j]);
}

// ---------------------------------------------------------------------------
// Kernel 2: attention for one (b, h, s): scores -> softmax -> ctx
// q tile 128x64, kv tile 64x64 (k == v). 8 waves, each owns 16 q-rows.
// q/kv tiles land in LDS via global_load_async_to_lds_b128 (ASYNCcnt).
// ---------------------------------------------------------------------------
__global__ __launch_bounds__(256) void attn_kernel(
    const __bf16* __restrict__ Q,   // [B*128, 768] bf16
    const __bf16* __restrict__ KV,  // [B*32*64, 768] bf16
    const float*  __restrict__ mask,// [B,32,64] f32
    __bf16* __restrict__ CTX) {     // [B*128*32, 768] bf16
  const int s = blockIdx.x, h = blockIdx.y, b = blockIdx.z;

  __shared__ __align__(16) unsigned char smem[65536];
  __bf16* lsQ   = (__bf16*)smem;              // 16KB: q tile [q][d]
  __bf16* lsAl  = (__bf16*)smem;              // 16KB: alpha (aliases lsQ)
  __bf16* lsKV  = (__bf16*)(smem + 16384);    //  8KB: kv [t][d]
  __bf16* lsKVT = (__bf16*)(smem + 24576);    //  8KB: kv^T [d][t]
  float*  lsS   = (float*)(smem + 32768);     // 32KB: scores per-wave 16x64

  const int tid  = threadIdx.x;
  const int wave = tid >> 5;

  { // stage q tile: 128 rows x 64 cols bf16 = 1024 x 16B chunks
    const __bf16* qsrc = Q + (size_t)b * 128 * 768 + h * 64;
#pragma unroll
    for (int i = 0; i < 4; ++i) {
      int idx = tid + i * 256;
      int r = idx >> 3, c8 = (idx & 7) * 8;
      cp16(&lsQ[idx * 8], &qsrc[(size_t)r * 768 + c8]);
    }
  }
  { // stage kv tile: 64 x 64 bf16 = 512 x 16B chunks
    const __bf16* ksrc = KV + (size_t)(b * 32 + s) * 64 * 768 + h * 64;
#pragma unroll
    for (int i = 0; i < 2; ++i) {
      int idx = tid + i * 256;
      int t = idx >> 3, d8 = (idx & 7) * 8;
      cp16(&lsKV[idx * 8], &ksrc[(size_t)t * 768 + d8]);
    }
  }
  wait_async0();
  __syncthreads();

  // build kv^T in LDS from the async-loaded kv tile
#pragma unroll
  for (int i = 0; i < 16; ++i) {
    int idx = tid + i * 256;
    int t = idx >> 6, d = idx & 63;
    lsKVT[d * 64 + t] = lsKV[t * 64 + d];
  }
  __syncthreads();

  // ---- scores[q,t] = sum_d q[q,d] * kv[t,d]  (B matrix = kv^T -> use lsKV)
  v8f sc[4] = {};
#pragma unroll
  for (int k0 = 0; k0 < 64; k0 += 32) {
    v16bf a = load_frag_a(&lsQ[(wave * 16) * 64 + k0], 64);
#pragma unroll
    for (int nt = 0; nt < 4; ++nt) {
      v16bf bf = load_frag_b(&lsKV[(nt * 16) * 64 + k0], 64);
      sc[nt] = wmma_bf16(a, bf, sc[nt]);
    }
  }
#pragma unroll
  for (int nt = 0; nt < 4; ++nt)
    store_frag_c_f32(&lsS[wave * 1024 + nt * 16], 64, sc[nt]);
  __syncthreads();

  // ---- row softmax over t=64 (scale 1/8, masked tokens get -10000)
  if (tid < 128) {
    float* srow = &lsS[(tid >> 4) * 1024 + (tid & 15) * 64];
    const float* mrow = mask + (size_t)(b * 32 + s) * 64;
    float mx = -3.4e38f;
#pragma unroll
    for (int t = 0; t < 64; ++t) {
      float v = srow[t] * 0.125f + mrow[t] * -10000.0f;
      srow[t] = v;
      mx = fmaxf(mx, v);
    }
    float sum = 0.0f;
#pragma unroll
    for (int t = 0; t < 64; ++t) {
      float e = __expf(srow[t] - mx);
      srow[t] = e;
      sum += e;
    }
    const float inv = 1.0f / sum;
    __bf16* arow = &lsAl[tid * 64];
#pragma unroll
    for (int t = 0; t < 64; ++t) arow[t] = f2bf(srow[t] * inv);
  }
  __syncthreads();

  // ---- ctx[q,d] = sum_t alpha[q,t] * kv[t,d]  (B matrix = kv -> use lsKVT)
  v8f cx[4] = {};
#pragma unroll
  for (int k0 = 0; k0 < 64; k0 += 32) {
    v16bf a = load_frag_a(&lsAl[(wave * 16) * 64 + k0], 64);
#pragma unroll
    for (int nt = 0; nt < 4; ++nt) {
      v16bf bf = load_frag_b(&lsKVT[(nt * 16) * 64 + k0], 64);
      cx[nt] = wmma_bf16(a, bf, cx[nt]);
    }
  }
  // ctx row index = (b*128 + q)*32 + s  -> q step is 32 rows of 768
  __bf16* cbase = CTX + ((size_t)(b * 128 + wave * 16) * 32 + s) * 768 + h * 64;
#pragma unroll
  for (int nt = 0; nt < 4; ++nt)
    store_frag_c_bf16(cbase + nt * 16, (size_t)32 * 768, cx[nt]);
}

// ---------------------------------------------------------------------------
// Kernel 3: OUT_f32[65536,768] = CTX_bf16 @ Wo_f32 + bo + ini_q[row/32]
// A tile (already bf16) staged via async-to-LDS; W tile converted f32->bf16.
// ---------------------------------------------------------------------------
__global__ __launch_bounds__(256) void out_gemm_kernel(
    const __bf16* __restrict__ CTX, const float* __restrict__ Wo,
    const float* __restrict__ bo, const float* __restrict__ resQ,
    float* __restrict__ OUT) {
  __shared__ __align__(16) __bf16 lsA[128 * 32];
  __shared__ __align__(16) __bf16 lsWT[64 * 32];

  const int tid  = threadIdx.x;
  const int wave = tid >> 5;
  const int wm   = wave >> 1;
  const int wn   = wave & 1;
  const int m0   = blockIdx.x * 128;
  const int n0   = blockIdx.y * 64;

  v8f acc[2][2] = {};
  for (int kk = 0; kk < 768; kk += 32) {
#pragma unroll
    for (int i = 0; i < 2; ++i) {        // A: 512 x 16B chunks, bf16 direct
      int idx = tid + i * 256;
      int r = idx >> 2, c8 = (idx & 3) * 8;
      cp16(&lsA[idx * 8], &CTX[(size_t)(m0 + r) * 768 + kk + c8]);
    }
#pragma unroll
    for (int i = 0; i < 2; ++i) {        // W: 512 float4s, transposed store
      int idx = tid + i * 256;
      int k = idx >> 4, n4 = (idx & 15) * 4;
      float4 w4 = *(const float4*)&Wo[(size_t)(kk + k) * 768 + n0 + n4];
      lsWT[(n4 + 0) * 32 + k] = f2bf(w4.x);
      lsWT[(n4 + 1) * 32 + k] = f2bf(w4.y);
      lsWT[(n4 + 2) * 32 + k] = f2bf(w4.z);
      lsWT[(n4 + 3) * 32 + k] = f2bf(w4.w);
    }
    if (kk + 32 < 768) {
      __builtin_prefetch(&CTX[(size_t)(m0 + (tid >> 1)) * 768 + kk + 32]);
      __builtin_prefetch(&Wo[(size_t)(kk + 32 + (tid >> 5)) * 768 + n0]);
    }
    wait_async0();
    __syncthreads();

    v16bf a0 = load_frag_a(&lsA[(wm * 32 + 0)  * 32], 32);
    v16bf a1 = load_frag_a(&lsA[(wm * 32 + 16) * 32], 32);
    v16bf b0 = load_frag_b(&lsWT[(wn * 32 + 0)  * 32], 32);
    v16bf b1 = load_frag_b(&lsWT[(wn * 32 + 16) * 32], 32);
    acc[0][0] = wmma_bf16(a0, b0, acc[0][0]);
    acc[0][1] = wmma_bf16(a0, b1, acc[0][1]);
    acc[1][0] = wmma_bf16(a1, b0, acc[1][0]);
    acc[1][1] = wmma_bf16(a1, b1, acc[1][1]);
    __syncthreads();
  }

  const int lane = tid & 31;
  const int nl   = lane & 15;
  const int mo   = (lane < 16) ? 0 : 8;
#pragma unroll
  for (int i = 0; i < 2; ++i)
#pragma unroll
    for (int j = 0; j < 2; ++j) {
      const int mg0 = m0 + wm * 32 + i * 16;
      const int ng  = n0 + wn * 32 + j * 16 + nl;
      const float bb = bo[ng];
#pragma unroll
      for (int r = 0; r < 8; ++r) {
        const int mg = mg0 + mo + r;     // mg = (b*128+q)*32 + s
        OUT[(size_t)mg * 768 + ng] =
            acc[i][j][r] + bb + resQ[(size_t)(mg >> 5) * 768 + ng];
      }
    }
}

// ---------------------------------------------------------------------------
// Kernel 4: in-place LayerNorm over 768 per row (one block of 256 per row)
// ---------------------------------------------------------------------------
__global__ __launch_bounds__(256) void ln_kernel(
    float* __restrict__ Y, const float* __restrict__ w,
    const float* __restrict__ bias) {
  const int row = blockIdx.x;
  float* y = Y + (size_t)row * 768;
  const int tid = threadIdx.x;

  float x0 = y[tid], x1 = y[tid + 256], x2 = y[tid + 512];

  __shared__ float red[8], red2[8];
  float s = x0 + x1 + x2;
#pragma unroll
  for (int off = 16; off > 0; off >>= 1) s += __shfl_xor(s, off, 32);
  if ((tid & 31) == 0) red[tid >> 5] = s;
  __syncthreads();
  float tot = 0.0f;
#pragma unroll
  for (int i = 0; i < 8; ++i) tot += red[i];
  const float mean = tot * (1.0f / 768.0f);

  const float d0 = x0 - mean, d1 = x1 - mean, d2 = x2 - mean;
  float s2 = d0 * d0 + d1 * d1 + d2 * d2;
#pragma unroll
  for (int off = 16; off > 0; off >>= 1) s2 += __shfl_xor(s2, off, 32);
  if ((tid & 31) == 0) red2[tid >> 5] = s2;
  __syncthreads();
  float tot2 = 0.0f;
#pragma unroll
  for (int i = 0; i < 8; ++i) tot2 += red2[i];
  const float inv = rsqrtf(tot2 * (1.0f / 768.0f) + 1e-12f);

  y[tid]       = d0 * inv * w[tid]       + bias[tid];
  y[tid + 256] = d1 * inv * w[tid + 256] + bias[tid + 256];
  y[tid + 512] = d2 * inv * w[tid + 512] + bias[tid + 512];
}

// ---------------------------------------------------------------------------
// Launch: q proj -> kv proj -> attention -> out proj (+bias+residual) -> LN
// ---------------------------------------------------------------------------
extern "C" void kernel_launch(void* const* d_in, const int* in_sizes, int n_in,
                              void* d_out, int out_size, void* d_ws,
                              size_t ws_size, hipStream_t stream) {
  (void)in_sizes; (void)n_in; (void)out_size; (void)ws_size;
  const float* ini_q = (const float*)d_in[0];   // [16,128,768]
  const float* ini_k = (const float*)d_in[1];   // [16,32,64,768]
  const float* mask  = (const float*)d_in[2];   // [16,32,64]
  const float* Wq    = (const float*)d_in[3];
  const float* Wkv   = (const float*)d_in[4];
  const float* Wo    = (const float*)d_in[5];
  const float* bo    = (const float*)d_in[6];
  const float* ln_w  = (const float*)d_in[7];
  const float* ln_b  = (const float*)d_in[8];
  float* out = (float*)d_out;                   // [16,128,32,768]

  char* ws = (char*)d_ws;
  __bf16* q_bf  = (__bf16*)ws;                            //  2048*768 bf16
  __bf16* kv_bf = (__bf16*)(ws + (size_t)3145728);        // 32768*768 bf16
  __bf16* ctx   = (__bf16*)(ws + (size_t)3145728 + 50331648); // 65536*768 bf16

  // Q = ini_q @ Wq        (M=2048)
  proj_gemm_kernel<<<dim3(16, 12), 256, 0, stream>>>(ini_q, Wq, q_bf);
  // KV = ini_k @ Wkv      (M=32768)
  proj_gemm_kernel<<<dim3(256, 12), 256, 0, stream>>>(ini_k, Wkv, kv_bf);
  // attention per (s, h, b)
  attn_kernel<<<dim3(32, 12, 16), 256, 0, stream>>>(q_bf, kv_bf, mask, ctx);
  // out = ctx @ Wo + bo + residual   (M=65536)
  out_gemm_kernel<<<dim3(512, 12), 256, 0, stream>>>(ctx, Wo, bo, ini_q, out);
  // in-place LayerNorm
  ln_kernel<<<65536, 256, 0, stream>>>(out, ln_w, ln_b);
}
